// EdgeFaceNet_59322088292459
// MI455X (gfx1250) — matrix operand order repaired and checked by reading, not compiled, not run on hardware
//
#include <hip/hip_runtime.h>
#include <hip/hip_bf16.h>

// ---------------------------------------------------------------------------
// EdgeFaceNet on MI455X (gfx1250): fused gather + 3-layer MLP + softmax.
// GEMM1/GEMM2 run on v_wmma_f32_16x16x32_bf16 (f32 accumulate); GEMM3 (N=2)
// runs on VALU float4 FMAs + wave32 shfl_xor for the 2-way softmax.
// ---------------------------------------------------------------------------

typedef __attribute__((ext_vector_type(16))) __bf16 v16bf;
typedef __attribute__((ext_vector_type(8)))  __bf16 v8bf;
typedef __attribute__((ext_vector_type(8)))  float  v8f;

#define WAVES_PER_BLOCK 4
#define BLOCK_THREADS   128
#define MAX_BLOCKS      640

// padded LDS strides (in elements) to spread rows across the 64 banks
#define A_STRIDE   136   // bf16, A tile rows: 16 x 128 (+8 pad)
#define W1T_STRIDE 136   // bf16, W1^T rows:   64 x 128 (+8 pad)
#define H1_STRIDE  72    // bf16, H1 rows:     16 x 64  (+8 pad)
#define W2T_STRIDE 72    // bf16, W2^T rows:   64 x 64  (+8 pad)
#define H2_STRIDE  68    // f32,  H2 rows:     16 x 64  (+4 pad)

__device__ __forceinline__ unsigned short f2bf_bits(float f) {
    unsigned u = __builtin_bit_cast(unsigned, f);
    unsigned r = (u + 0x7FFFu + ((u >> 16) & 1u)) >> 16;   // round-to-nearest-even
    return (unsigned short)r;
}

__device__ __forceinline__ __bf16 f2bf(float f) {
    unsigned short h = f2bf_bits(f);
    return __builtin_bit_cast(__bf16, h);
}

__device__ __forceinline__ unsigned packbf2(float a, float b) {
    return (unsigned)f2bf_bits(a) | ((unsigned)f2bf_bits(b) << 16);
}

// Build a 16-element bf16 fragment from two 16-byte LDS chunks
// (elements 0..7 -> VGPR0..3, elements 8..15 -> VGPR4..7).
__device__ __forceinline__ v16bf load_frag16(const __bf16* p0, const __bf16* p1) {
    v8bf lo = *(const v8bf*)p0;
    v8bf hi = *(const v8bf*)p1;
    v16bf r;
#pragma unroll
    for (int i = 0; i < 8; ++i) { r[i] = lo[i]; r[i + 8] = hi[i]; }
    return r;
}

__global__ __launch_bounds__(BLOCK_THREADS)
void edgeface_fused_kernel(const float* __restrict__ topo,
                           const long long* __restrict__ esid,
                           const float* __restrict__ W1, const float* __restrict__ b1,
                           const float* __restrict__ W2, const float* __restrict__ b2,
                           const float* __restrict__ W3, const float* __restrict__ b3,
                           float* __restrict__ out, int rows)
{
    // ---- static LDS (~54 KB) ----
    __shared__ __align__(16) __bf16 sW1T[64 * W1T_STRIDE];          // 17408 B
    __shared__ __align__(16) __bf16 sW2T[64 * W2T_STRIDE];          //  9216 B
    __shared__ __align__(16) float  sW3T[2 * 64];                   //   512 B
    __shared__ float sB1[64], sB2[64], sB3[2];
    __shared__ __align__(16) __bf16 sA [WAVES_PER_BLOCK][16 * A_STRIDE];  // 4 x 4352 B (reused as f32 H2)
    __shared__ __align__(16) __bf16 sH1[WAVES_PER_BLOCK][16 * H1_STRIDE]; // 4 x 2304 B

    const int tid = threadIdx.x;

    // ---- stage weights (f32 global -> bf16 transposed LDS), biases ----
    for (int idx = tid; idx < 64 * 128; idx += BLOCK_THREADS) {
        const int n = idx & 63, k = idx >> 6;                       // W1 is [k=128][n=64]
        sW1T[n * W1T_STRIDE + k] = f2bf(W1[k * 64 + n]);
    }
    for (int idx = tid; idx < 64 * 64; idx += BLOCK_THREADS) {
        const int n = idx & 63, k = idx >> 6;                       // W2 is [k=64][n=64]
        sW2T[n * W2T_STRIDE + k] = f2bf(W2[k * 64 + n]);
    }
    for (int idx = tid; idx < 128; idx += BLOCK_THREADS) {
        const int o = idx & 1, k = idx >> 1;                        // W3 is [k=64][o=2]
        sW3T[o * 64 + k] = W3[k * 2 + o];
    }
    if (tid < 64) { sB1[tid] = b1[tid]; sB2[tid] = b2[tid]; }
    if (tid < 2)  { sB3[tid] = b3[tid]; }
    __syncthreads();

    const int wave = tid >> 5;
    const int lane = tid & 31;
    const int lr = lane & 15;   // row (A) / col (B,D) index within 16
    const int lh = lane >> 4;   // lane-half selector for K-chunk placement

    __bf16* A  = &sA[wave][0];
    __bf16* H1 = &sH1[wave][0];
    float*  H2 = (float*)&sA[wave][0];   // A tile is dead once GEMM1 is done

    const int tiles = (rows + 15) >> 4;

    for (int tile = blockIdx.x * WAVES_PER_BLOCK + wave; tile < tiles;
         tile += gridDim.x * WAVES_PER_BLOCK) {

        // ============ gather: 16 rows x (64 self | 64 neighbor) -> bf16 A ============
        {
            const int i   = lane >> 1;        // row within tile
            const int seg = lane & 1;         // 0 = edge feat, 1 = neighbor feat
            const int g   = tile * 16 + i;    // global row = edge*3 + n
            if (g < rows) {
                const int e = g / 3;
                const int n = g - 3 * e;
                long long x = esid[3 * e + 0];
                long long y = esid[3 * e + 1];
                long long z = esid[3 * e + 2];
                if (seg) {                    // CUBE_NEIGHBOR rows, clipped at 0
                    if (n == 0)      x = (x > 0) ? x - 1 : 0;
                    else if (n == 1) y = (y > 0) ? y - 1 : 0;
                    else             z = (z > 0) ? z - 1 : 0;
                }
                const size_t base = (((size_t)x * 128 + (size_t)y) * 128 + (size_t)z) * 64;
                const float4* src = (const float4*)(topo + base);
                uint2* dst = (uint2*)(A + i * A_STRIDE + seg * 64);
#pragma unroll
                for (int j = 0; j < 16; ++j) {
                    const float4 f = src[j];
                    uint2 p;
                    p.x = packbf2(f.x, f.y);
                    p.y = packbf2(f.z, f.w);
                    dst[j] = p;
                }
            }
        }
        asm volatile("s_wait_dscnt 0" ::: "memory");

        // ============ GEMM1: (16x128 bf16) x (128x64 bf16) -> f32 ============
        v8f acc[4];
#pragma unroll
        for (int nt = 0; nt < 4; ++nt) acc[nt] = v8f{};
        {
            const __bf16* arow = A + lr * A_STRIDE;
#pragma unroll
            for (int ks = 0; ks < 4; ++ks) {
                // A layout: lane holds row lr, K chunks [ks*32+lh*8 .. +8) and +16
                const __bf16* ap = arow + ks * 32 + lh * 8;
                const v16bf a = load_frag16(ap, ap + 16);
#pragma unroll
                for (int nt = 0; nt < 4; ++nt) {
                    // B layout: lane holds col lr(+16*nt), K chunks [ks*32+lh*8 .. +8) and +16
                    const __bf16* bp = sW1T + (lr + nt * 16) * W1T_STRIDE + ks * 32 + lh * 8;
                    const v16bf b = load_frag16(bp, bp + 16);
                    acc[nt] = __builtin_amdgcn_wmma_f32_16x16x32_bf16(
                        false, a, false, b, (short)0, acc[nt], false, false);
                }
            }
        }
        // bias + relu -> H1 (bf16), D layout: (m = v + 8*lh, n = lr + 16*nt)
#pragma unroll
        for (int nt = 0; nt < 4; ++nt) {
            const int n = lr + nt * 16;
            const float bias = sB1[n];
#pragma unroll
            for (int v = 0; v < 8; ++v) {
                const int m = v + 8 * lh;
                float h = acc[nt][v] + bias;
                h = (h > 0.f) ? h : 0.f;
                H1[m * H1_STRIDE + n] = f2bf(h);
            }
        }
        asm volatile("s_wait_dscnt 0" ::: "memory");

        // ============ GEMM2: (16x64 bf16) x (64x64 bf16) -> f32 ============
#pragma unroll
        for (int nt = 0; nt < 4; ++nt) acc[nt] = v8f{};
        {
            const __bf16* hrow = H1 + lr * H1_STRIDE;
#pragma unroll
            for (int ks = 0; ks < 2; ++ks) {
                const __bf16* ap = hrow + ks * 32 + lh * 8;
                const v16bf a = load_frag16(ap, ap + 16);
#pragma unroll
                for (int nt = 0; nt < 4; ++nt) {
                    const __bf16* bp = sW2T + (lr + nt * 16) * W2T_STRIDE + ks * 32 + lh * 8;
                    const v16bf b = load_frag16(bp, bp + 16);
                    acc[nt] = __builtin_amdgcn_wmma_f32_16x16x32_bf16(
                        false, a, false, b, (short)0, acc[nt], false, false);
                }
            }
        }
        // bias + relu -> H2 (f32, aliases dead A tile)
#pragma unroll
        for (int nt = 0; nt < 4; ++nt) {
            const int n = lr + nt * 16;
            const float bias = sB2[n];
#pragma unroll
            for (int v = 0; v < 8; ++v) {
                const int m = v + 8 * lh;
                float h = acc[nt][v] + bias;
                H2[m * H2_STRIDE + n] = (h > 0.f) ? h : 0.f;
            }
        }
        asm volatile("s_wait_dscnt 0" ::: "memory");

        // ============ GEMM3 (N=2, VALU) + 2-way softmax ============
        {
            const int r = lr;        // row within tile (lanes 0-15 / 16-31 duplicate rows)
            const int o = lh;        // output channel 0/1
            float vacc = sB3[o];
            const float4* h4 = (const float4*)(H2 + r * H2_STRIDE);
            const float4* w4 = (const float4*)(sW3T + o * 64);
#pragma unroll
            for (int j = 0; j < 16; ++j) {
                const float4 h = h4[j];
                const float4 w = w4[j];
                vacc = fmaf(h.x, w.x, vacc);
                vacc = fmaf(h.y, w.y, vacc);
                vacc = fmaf(h.z, w.z, vacc);
                vacc = fmaf(h.w, w.w, vacc);
            }
            const float vo = __shfl_xor(vacc, 16, 32);   // partner logit (wave32)
            const float mx = fmaxf(vacc, vo);
            const float ea = __expf(vacc - mx);
            const float eb = __expf(vo - mx);
            const float p  = ea / (ea + eb);
            const int g = tile * 16 + r;
            if (g < rows) {
                const int e  = g / 3;
                const int n3 = g - 3 * e;
                out[(size_t)e * 6 + o * 3 + n3] = p;     // out[e][o][n]
            }
        }
    }
}

extern "C" void kernel_launch(void* const* d_in, const int* in_sizes, int n_in,
                              void* d_out, int out_size, void* d_ws, size_t ws_size,
                              hipStream_t stream) {
    (void)n_in; (void)out_size; (void)d_ws; (void)ws_size;
    const float*     topo = (const float*)d_in[0];
    const long long* esid = (const long long*)d_in[1];   // reference dtype is int64
    const float* W1 = (const float*)d_in[2];
    const float* b1 = (const float*)d_in[3];
    const float* W2 = (const float*)d_in[4];
    const float* b2 = (const float*)d_in[5];
    const float* W3 = (const float*)d_in[6];
    const float* b3 = (const float*)d_in[7];
    float* out = (float*)d_out;

    const int rows  = in_sizes[1];               // N_es * 3 (flat esid count)
    const int tiles = (rows + 15) >> 4;
    int blocks = (tiles + WAVES_PER_BLOCK - 1) / WAVES_PER_BLOCK;
    if (blocks > MAX_BLOCKS) blocks = MAX_BLOCKS;
    if (blocks < 1) blocks = 1;

    hipLaunchKernelGGL(edgeface_fused_kernel, dim3(blocks), dim3(BLOCK_THREADS), 0, stream,
                       topo, esid, W1, b1, W2, b2, W3, b3, out, rows);
}
